// CalculateAttention_86354612453682
// MI455X (gfx1250) — matrix operand order repaired and checked by reading, not compiled, not run on hardware
//
#include <hip/hip_runtime.h>
#include <stdint.h>

// CDNA5 / gfx1250 flash-attention forward (transposed formulation).
// B=4, H=16, S=2048, D=64, fp32 in/out, bool mask [B,1,S,S] (true = masked).
//
// Per block: 8 waves x 16 query rows = 128 rows, one (b,h).
// Per 64-wide KV tile:
//   S^T = K_tile . Qs^T     (A = K from LDS, B = pre-scaled Q in regs)  8 WMMAs
//   O^T += V^T_tile . P^T   (A = V^T from LDS, B = P in regs)           8 WMMAs
// Softmax in base-2 with scale folded into Q; row stats per-lane scalars.

typedef __bf16 bf16_t;
typedef __attribute__((ext_vector_type(16))) __bf16 v16bf;
typedef __attribute__((ext_vector_type(8)))  __bf16 v8bf;
typedef __attribute__((ext_vector_type(8)))  float  v8f;

#define NB    4
#define NH    16
#define SEQ   2048
#define DH    64
#define QTILE 128      // query rows per block (8 waves x 16 rows)
#define KVT   64       // kv columns per iteration
#define PITCH 72       // LDS row pitch in bf16 (144 B, multiple of 16 B)

// log2(e)/sqrt(64) folded into Q; mask bias = -1e9/sqrt(64)*log2(e)
#define SCALE2 0.18033688011112042f
#define MBIAS  (-1.8033688011112042e8f)

#if __has_builtin(__builtin_amdgcn_exp2f)
#define EXP2F __builtin_amdgcn_exp2f
#else
#define EXP2F exp2f
#endif

// round-to-nearest-away f32 -> bf16 (1 VALU)
__device__ __forceinline__ bf16_t f2bf(float f) {
    uint32_t u = __builtin_bit_cast(uint32_t, f) + 0x8000u;
    return __builtin_bit_cast(bf16_t, (uint16_t)(u >> 16));
}

// pack two f32 into bf16x2: 2 adds + 1 v_perm_b32
__device__ __forceinline__ uint32_t packbf2(float a, float b) {
    uint32_t au = __builtin_bit_cast(uint32_t, a) + 0x8000u;
    uint32_t bu = __builtin_bit_cast(uint32_t, b) + 0x8000u;
    // result bytes (LSB..MSB): au.b2, au.b3, bu.b2, bu.b3
    return __builtin_amdgcn_perm(bu, au, 0x07060302u);
}

union V16U { v16bf v; v8bf h[2]; uint32_t u[8]; };

__global__ __launch_bounds__(256)
void fa_fwd_bf16wmma(const float* __restrict__ Q, const float* __restrict__ K,
                     const float* __restrict__ V, const uint8_t* __restrict__ M,
                     float* __restrict__ O)
{
    __shared__ bf16_t Ks[KVT][PITCH];       // K tile, row-major [kv][d]
    __shared__ bf16_t Vt[DH][PITCH];        // V tile, transposed [d][kv]

    const int tid  = threadIdx.x;
    const int wave = tid >> 5;
    const int lane = tid & 31;
    const int hb   = (lane >> 4) & 1;       // which 16-lane half
    const int l15  = lane & 15;

    const int b  = blockIdx.z;
    const int h  = blockIdx.y;
    const int q0 = blockIdx.x * QTILE;

    const size_t headoff = ((size_t)b * NH + h) * (size_t)SEQ * DH;
    const float* Qg = Q + headoff;
    const float* Kg = K + headoff;
    const float* Vg = V + headoff;
    float*       Og = O + headoff;

    const int qrow = q0 + wave * 16;        // first query row of this wave
    const int qme  = qrow + l15;            // this lane's query row (both halves)
    const uint8_t* Mrow = M + (size_t)b * SEQ * SEQ + (size_t)qme * SEQ;

    // ---- pre-scaled Q B-fragments in registers (bf16), 2 chunks over D ----
    // B 32x16 layout: lane holds N = l15 (q col), K(d) = 32c + 16*hb + i
    V16U qb[2];
    {
        const float* qp = Qg + (size_t)qme * DH;
#pragma unroll
        for (int c = 0; c < 2; ++c) {
            const int base = 32 * c + 16 * hb;
#pragma unroll
            for (int i = 0; i < 8; ++i)
                qb[c].u[i] = packbf2(qp[base + 2*i] * SCALE2,
                                     qp[base + 2*i + 1] * SCALE2);
        }
    }

    // ---- online-softmax state (scalar per lane: this lane's q row) ----
    v8f o[4];
    {
        v8f zero = {};
#pragma unroll
        for (int t = 0; t < 4; ++t) o[t] = zero;
    }
    float m = -3.0e38f;   // base-2 running max
    float l = 0.0f;       // running denominator

    for (int kv0 = 0; kv0 < SEQ; kv0 += KVT) {
        // ---- cooperative load: K tile row-major (packed), V tile transposed ----
        {
            const int row = tid & 63;             // kv row
            const int cg  = (tid >> 6) * 16;      // d column group
            const float* kp = Kg + (size_t)(kv0 + row) * DH + cg;
            const float* vp = Vg + (size_t)(kv0 + row) * DH + cg;
#pragma unroll
            for (int i = 0; i < 8; ++i)
                *(uint32_t*)&Ks[row][cg + 2*i] = packbf2(kp[2*i], kp[2*i + 1]);
#pragma unroll
            for (int i = 0; i < 16; ++i) Vt[cg + i][row] = f2bf(vp[i]);
            if (kv0 + KVT < SEQ) {
                __builtin_prefetch(kp + (size_t)KVT * DH, 0, 0);
                __builtin_prefetch(vp + (size_t)KVT * DH, 0, 0);
            }
        }

        // ---- mask words: 8 consecutive bytes per tile for this lane's row ----
        uint2 mw[4];
        {
            const uint8_t* mp = Mrow + kv0 + 8 * hb;
#pragma unroll
            for (int t = 0; t < 4; ++t) mw[t] = *(const uint2*)(mp + 16 * t);
        }
        __syncthreads();

        // ---- S^T = K . Qs^T : tile t covers kv rows 16t..16t+15 ----
        v8f s[4];
#pragma unroll
        for (int t = 0; t < 4; ++t) {
            v8f acc = {};
#pragma unroll
            for (int c = 0; c < 2; ++c) {
                // A 16x32 layout: lane M = l15 (kv), runs K = 16*hb..16*hb+15 (contiguous)
                V16U ak;
                const bf16_t* kr = &Ks[16 * t + l15][32 * c + 16 * hb];
                ak.h[0] = *(const v8bf*)(kr);
                ak.h[1] = *(const v8bf*)(kr + 8);
                acc = __builtin_amdgcn_wmma_f32_16x16x32_bf16(
                          false, ak.v, false, qb[c].v, (short)0, acc, false, false);
            }
            s[t] = acc;
        }

        // ---- mask bias (scale already folded in) ; row max over 32 kv vals ----
        float mx = -3.0e38f;
#pragma unroll
        for (int t = 0; t < 4; ++t) {
#pragma unroll
            for (int j = 0; j < 8; ++j) {
                const uint32_t w = (j < 4) ? mw[t].x : mw[t].y;
                const float mkf = (float)((w >> (8 * (j & 3))) & 0xffu);
                const float z = fmaf(mkf, MBIAS, s[t][j]);
                s[t][j] = z;
                mx = fmaxf(mx, z);
            }
        }
        mx = fmaxf(mx, __shfl_xor(mx, 16, 32));   // combine kv halves
        const float mn = fmaxf(m, mx);
        const float alpha = EXP2F(m - mn);
        m = mn;

        // ---- p = 2^(z-m), pack bf16 pairs, row sum ----
        float rs = 0.0f;
        uint32_t pb[4][4];
#pragma unroll
        for (int t = 0; t < 4; ++t) {
#pragma unroll
            for (int w = 0; w < 4; ++w) {
                const float p0 = EXP2F(s[t][2 * w]     - m);
                const float p1 = EXP2F(s[t][2 * w + 1] - m);
                rs += p0 + p1;
                pb[t][w] = packbf2(p0, p1);
            }
        }
        rs += __shfl_xor(rs, 16, 32);
        l = l * alpha + rs;
#pragma unroll
        for (int t = 0; t < 4; ++t)
#pragma unroll
            for (int j = 0; j < 8; ++j) o[t][j] *= alpha;

        // ---- O^T += V^T . P^T ----
#pragma unroll
        for (int c = 0; c < 2; ++c) {
            // Build P^T B-fragment: lane needs tile 2c+hb, both kv halves.
            V16U pf;
#pragma unroll
            for (int w = 0; w < 4; ++w) {
                const uint32_t keep = hb ? pb[2 * c + 1][w] : pb[2 * c][w];
                const uint32_t send = hb ? pb[2 * c][w]     : pb[2 * c + 1][w];
                const uint32_t recv = (uint32_t)__shfl_xor((int)send, 16, 32);
                pf.u[w]     = hb ? recv : keep;   // elements 0..7  (kv half 0)
                pf.u[4 + w] = hb ? keep : recv;   // elements 8..15 (kv half 1)
            }
#pragma unroll
            for (int t = 0; t < 4; ++t) {
                // A: lane M = l15 (d row 16t+l15), contraction kv chunk c
                V16U av;
                const bf16_t* vr = &Vt[16 * t + l15][32 * c + 16 * hb];
                av.h[0] = *(const v8bf*)(vr);
                av.h[1] = *(const v8bf*)(vr + 8);
                o[t] = __builtin_amdgcn_wmma_f32_16x16x32_bf16(
                           false, av.v, false, pf.v, (short)0, o[t], false, false);
            }
        }
        __syncthreads();
    }

    // ---- normalize + store: lane owns row qme, d = 16t + 8hb + 0..7 ----
    const float inv = 1.0f / l;
    float* orow = Og + (size_t)qme * DH + 8 * hb;
#pragma unroll
    for (int t = 0; t < 4; ++t) {
        float4 lo, hi;
        lo.x = o[t][0] * inv; lo.y = o[t][1] * inv;
        lo.z = o[t][2] * inv; lo.w = o[t][3] * inv;
        hi.x = o[t][4] * inv; hi.y = o[t][5] * inv;
        hi.z = o[t][6] * inv; hi.w = o[t][7] * inv;
        *(float4*)(orow + 16 * t)     = lo;
        *(float4*)(orow + 16 * t + 4) = hi;
    }
}

extern "C" void kernel_launch(void* const* d_in, const int* in_sizes, int n_in,
                              void* d_out, int out_size, void* d_ws, size_t ws_size,
                              hipStream_t stream) {
    (void)in_sizes; (void)n_in; (void)out_size; (void)d_ws; (void)ws_size;
    const float*   Q = (const float*)d_in[0];
    const float*   K = (const float*)d_in[1];
    const float*   V = (const float*)d_in[2];
    const uint8_t* M = (const uint8_t*)d_in[3];   // jnp bool -> 1 byte/elem
    float*         O = (float*)d_out;

    dim3 grid(SEQ / QTILE, NH, NB);
    dim3 block(256);
    fa_fwd_bf16wmma<<<grid, block, 0, stream>>>(Q, K, V, M, O);
}
